// ETA_47485158425279
// MI455X (gfx1250) — matrix operand order repaired and checked by reading, not compiled, not run on hardware
//
#include <hip/hip_runtime.h>

// ---------------- CDNA5 WMMA types / helpers ----------------
typedef __attribute__((ext_vector_type(16))) _Float16 v16h;
typedef __attribute__((ext_vector_type(8)))  _Float16 v8h;
typedef __attribute__((ext_vector_type(4)))  _Float16 v4h;
typedef __attribute__((ext_vector_type(8)))  float    v8f;

#define Bn    512
#define Sn    2048
#define KTOP  128
#define En    64
#define An    128
#define Hn    4
#define DKn   32
#define ODn   64
#define FOHn  15
#define FMHn  4
#define FSPn  4
#define LSPn  50
#define VOHn  10001
#define VMHn  100001
#define VSPn  10001
#define D0n   1728
#define NROWS_MH (FMHn*VMHn)

static __device__ __forceinline__ v8f wmma16(v16h a, v16h b, v8f c) {
  // D = A(16x32) * B(32x16) + C, f32 accum, wave32
  return __builtin_amdgcn_wmma_f32_16x16x32_f16(false, a, false, b, (short)0, c,
                                                false, false);
}

// A-fragment (16x32 f16, row-major tile in LDS, leading dim lda).
// lane (hl=lane>>4, r=lane&15): elems 0..15 -> K 16*hl+0..15 (matches swizzled B).
// One contiguous 32B run -> two ds_load_b128.
static __device__ __forceinline__ v16h frag_a(const _Float16* tile, int lda, int lane) {
  const int hl = (lane >> 4) & 1, r = lane & 15;
  const _Float16* p = tile + r * lda + 16 * hl;
  const v8h lo = *(const v8h*)(p);
  const v8h hi = *(const v8h*)(p + 8);
  return __builtin_shufflevector(lo, hi, 0, 1, 2, 3, 4, 5, 6, 7,
                                 8, 9, 10, 11, 12, 13, 14, 15);
}

// Swizzled-address for B tiles staged in fragment order.
// A 32(K) x 128(N) chunk = 8 column-tiles of 512 halfs; within a tile the
// fragment for lane L is 16 contiguous halfs: lane = (n&15) + 16*(k>>4),
// elem j = k&15.
static __device__ __forceinline__ int swz_b(int k, int n) {
  return ((n >> 4) << 9) + ((((n & 15) + ((k >> 4) << 4)) << 4) | (k & 15));
}

// B-fragment load from a swizzled tile: two contiguous ds_load_b128.
static __device__ __forceinline__ v16h frag_b_sw(const _Float16* tileSw, int lane) {
  const _Float16* p = tileSw + lane * 16;
  const v8h lo = *(const v8h*)(p);
  const v8h hi = *(const v8h*)(p + 8);
  return __builtin_shufflevector(lo, hi, 0, 1, 2, 3, 4, 5, 6, 7,
                                 8, 9, 10, 11, 12, 13, 14, 15);
}

// ---------------- K1: LSH codes for every multi-hot vocab row ----------------
__global__ __launch_bounds__(256) void k_hash_mh(const float* __restrict__ mhT,
                                                 const float* __restrict__ hw,
                                                 unsigned int* __restrict__ codes) {
  const int wave = (blockIdx.x * 256 + threadIdx.x) >> 5;
  const int lane = threadIdx.x & 31;
  if (wave >= NROWS_MH) return;
  const float* row = mhT + (size_t)wave * En;
  float s = 0.f;
#pragma unroll 8
  for (int e = 0; e < En; ++e) s += row[e] * hw[e * 32 + lane];
  unsigned int bit = (s > 0.f) ? (1u << lane) : 0u;
#pragma unroll
  for (int off = 16; off >= 1; off >>= 1) bit |= __shfl_xor(bit, off, 32);
  if (lane == 0) codes[wave] = bit;
}

// ---------------- K2: LSH codes for the 4x512 target rows ----------------
__global__ __launch_bounds__(256) void k_hash_tgt(const float* __restrict__ ohT,
                                                  const float* __restrict__ hw,
                                                  const int* __restrict__ ohIds,
                                                  unsigned int* __restrict__ codes) {
  const int wave = (blockIdx.x * 256 + threadIdx.x) >> 5;
  const int lane = threadIdx.x & 31;
  if (wave >= FMHn * Bn) return;
  const int f = wave >> 9, b = wave & (Bn - 1);
  const int id = ohIds[b * FOHn + f];
  const float* row = ohT + ((size_t)f * VOHn + id) * En;
  float s = 0.f;
#pragma unroll 8
  for (int e = 0; e < En; ++e) s += row[e] * hw[e * 32 + lane];
  unsigned int bit = (s > 0.f) ? (1u << lane) : 0u;
#pragma unroll
  for (int off = 16; off >= 1; off >>= 1) bit |= __shfl_xor(bit, off, 32);
  if (lane == 0) codes[wave] = bit;
}

// ---------------- K3: exact stable top-128 by hamming distance ----------------
__global__ __launch_bounds__(256) void k_topk(const int* __restrict__ mhIds,
                                              const unsigned int* __restrict__ mhCodes,
                                              const unsigned int* __restrict__ tgtCodes,
                                              int* __restrict__ topkOut) {
  __shared__ unsigned short hcnt[34 * 256];
  __shared__ unsigned int bbase[34];
  const int tid = threadIdx.x;
  const int probe = blockIdx.x;          // f*512 + b
  const int f = probe >> 9;
  const unsigned int tc = tgtCodes[probe];
  const int* ids = mhIds + (size_t)probe * Sn;

  for (int i = tid; i < 34 * 256; i += 256) hcnt[i] = 0;
  __syncthreads();

  int mb[8];
#pragma unroll
  for (int i = 0; i < 8; ++i) {
    const int s = tid * 8 + i;
    const int id = ids[s];
    int bkt;
    if (id < 0) bkt = 33;
    else bkt = __popc(mhCodes[(size_t)f * VMHn + id] ^ tc);
    mb[i] = bkt;
    hcnt[bkt * 256 + tid] = (unsigned short)(hcnt[bkt * 256 + tid] + 1);
  }
  __syncthreads();

  if (tid < 34) {  // exclusive scan across threads, per bucket
    unsigned int sum = 0;
    for (int t = 0; t < 256; ++t) {
      const unsigned int c = hcnt[tid * 256 + t];
      hcnt[tid * 256 + t] = (unsigned short)sum;
      sum += c;
    }
    bbase[tid] = sum;
  }
  __syncthreads();
  if (tid == 0) {  // exclusive scan of bucket totals
    unsigned int acc = 0;
    for (int k = 0; k < 34; ++k) { const unsigned int c = bbase[k]; bbase[k] = acc; acc += c; }
  }
  __syncthreads();

  int* out = topkOut + (size_t)probe * KTOP;
#pragma unroll
  for (int i = 0; i < 8; ++i) {
    const int bkt = mb[i];
    int occ = 0;
#pragma unroll
    for (int j = 0; j < 8; ++j) occ += (j < i && mb[j] == bkt) ? 1 : 0;
    const unsigned int rank = bbase[bkt] + hcnt[bkt * 256 + tid] + occ;
    if (rank < KTOP) out[rank] = tid * 8 + i;
  }
}

// ---------------- K4: one-hot features into concat buffer ----------------
__global__ void k_oh_feats(const float* __restrict__ ohT, const int* __restrict__ ohIds,
                           float* __restrict__ feats) {
  const int t = blockIdx.x * 256 + threadIdx.x;
  if (t >= Bn * FOHn * En) return;
  const int d = t & 63;
  const int f = (t >> 6) % FOHn;
  const int b = t / (FOHn * En);
  const int id = ohIds[b * FOHn + f];
  feats[(size_t)b * D0n + f * En + d] = ohT[((size_t)f * VOHn + id) * En + d];
}

// ---------------- K5: special (sum-combined) features ----------------
__global__ void k_sp_feats(const float* __restrict__ spT, const int* __restrict__ spIds,
                           float* __restrict__ feats) {
  const int t = blockIdx.x * 256 + threadIdx.x;
  if (t >= FSPn * Bn * En) return;
  const int d = t & 63;
  const int b = (t >> 6) & (Bn - 1);
  const int f = t >> 15;
  const int* ids = spIds + ((size_t)f * Bn + b) * LSPn;
  float acc = 0.f;
  for (int l = 0; l < LSPn; ++l) {
    const int id = ids[l];
    if (id >= 0) acc += spT[((size_t)f * VSPn + id) * En + d];
  }
  feats[(size_t)b * D0n + FOHn * En + f * En + d] = acc;
}

// ---------------- K6: fused attention (short & long), WMMA K/V projections ----
// grid.x = 4096: bit11 = type (0=short first-K, 1=long topk), bits10:9 = field,
// bits8:0 = batch. 256 threads = 8 waves; 96 KB dynamic LDS.
__global__ __launch_bounds__(256) void k_attn(
    const float* __restrict__ ohT, const float* __restrict__ mhT,
    const int* __restrict__ ohIds, const int* __restrict__ mhIds,
    const int* __restrict__ topkIdx,
    const float* sqW, const float* sqb, const float* skW, const float* skb,
    const float* svW, const float* svb, const float* soW, const float* sob,
    const float* lqW, const float* lqb, const float* lkW, const float* lkb,
    const float* lvW, const float* lvb, const float* loW, const float* lob,
    float* __restrict__ feats) {
  extern __shared__ char smem[];
  _Float16* s_seq = (_Float16*)smem;               // 128 x 64  f16 (16 KB), row-major
  _Float16* s_w   = (_Float16*)(smem + 16384);     // 64 x 128  f16 (16 KB), SWIZZLED
  float*    s_x   = (float*)(smem + 32768);        // 128 x 128 f32 (64 KB)
  __shared__ __align__(16) float s_q[An];
  __shared__ __align__(16) float s_sc[Hn * KTOP];
  __shared__ __align__(16) float s_out[An];
  __shared__ __align__(16) float s_tgt[En];
  __shared__ int s_valid[KTOP];

  const int tid = threadIdx.x, lane = tid & 31, wave = tid >> 5;
  const int gb = blockIdx.x;
  const int type = gb >> 11;
  const int f = (gb >> 9) & 3;
  const int b = gb & (Bn - 1);
  const int probe = gb & 2047;                     // f*512 + b

  const float* qW = (type ? lqW : sqW) + (size_t)f * En * An;
  const float* qb = (type ? lqb : sqb) + f * An;
  const float* kW = (type ? lkW : skW) + (size_t)f * En * An;
  const float* kb = (type ? lkb : skb) + f * An;
  const float* vW = (type ? lvW : svW) + (size_t)f * En * An;
  const float* vb = (type ? lvb : svb) + f * An;
  const float* oW = (type ? loW : soW) + (size_t)f * An * ODn;
  const float* ob = (type ? lob : sob) + f * ODn;

  // target embedding row
  if (tid < En) {
    const int oid = ohIds[b * FOHn + f];
    s_tgt[tid] = ohT[((size_t)f * VOHn + oid) * En + tid];
  }
  // gather 128 sequence rows (two threads per row, float4 loads, f32->f16)
  {
    const int row = tid >> 1;
    const int off = (tid & 1) * 32;
    const int s = (type == 0) ? row : topkIdx[(size_t)probe * KTOP + row];
    int id = mhIds[(size_t)probe * Sn + s];
    const int valid = id >= 0;
    if (!valid) id = 0;
    const float4* src = (const float4*)(mhT + ((size_t)f * VMHn + id) * En + off);
#pragma unroll
    for (int j = 0; j < 8; ++j) {
      const float4 v = src[j];
      v4h h;
      h[0] = (_Float16)v.x; h[1] = (_Float16)v.y;
      h[2] = (_Float16)v.z; h[3] = (_Float16)v.w;
      *(v4h*)(s_seq + row * En + off + j * 4) = h;
    }
    if ((tid & 1) == 0) s_valid[row] = valid;
  }
  // stage kW (64x128) as f16, swizzled into fragment order.
  // chunk = e>>5 selects the K=0..31 / K=32..63 half (4096 halfs each).
#pragma unroll
  for (int i = tid * 4; i < En * An; i += 256 * 4) {
    const int e = i >> 7, a = i & 127;
    const float4 w4 = *(const float4*)(kW + i);
    _Float16* dst = s_w + ((e >> 5) << 12);
    const int ek = e & 31;
    dst[swz_b(ek, a + 0)] = (_Float16)w4.x;
    dst[swz_b(ek, a + 1)] = (_Float16)w4.y;
    dst[swz_b(ek, a + 2)] = (_Float16)w4.z;
    dst[swz_b(ek, a + 3)] = (_Float16)w4.w;
  }
  __syncthreads();

  // q = tgt @ qW + qb   (tiny: VALU)
  if (tid < An) {
    float acc = qb[tid];
#pragma unroll 8
    for (int e = 0; e < En; ++e) acc += s_tgt[e] * qW[e * An + tid];
    s_q[tid] = acc;
  }

  // K = seq @ kW + kb via WMMA; wave owns rows 16*wave..16*wave+15
  const int rowbase = 16 * wave;
  const v16h a0 = frag_a(s_seq + rowbase * En + 0, En, lane);
  const v16h a1 = frag_a(s_seq + rowbase * En + 32, En, lane);
  const int hl = lane >> 4, nn = lane & 15;
#pragma unroll
  for (int ct = 0; ct < 8; ++ct) {
    v8f acc = {};
    acc = wmma16(a0, frag_b_sw(s_w + ct * 512, lane), acc);
    acc = wmma16(a1, frag_b_sw(s_w + 4096 + ct * 512, lane), acc);
    const int c = 16 * ct + nn;
    const float bias = kb[c];
#pragma unroll
    for (int i = 0; i < 8; ++i) s_x[(rowbase + i + 8 * hl) * An + c] = acc[i] + bias;
  }
  __syncthreads();

  // scores[h][s] = q_h . k_s / sqrt(32), masked (float4 LDS reads)
  const float scale = 0.17677669529663687f;
  for (int it = tid; it < Hn * KTOP; it += 256) {
    const int h = it >> 7, s = it & (KTOP - 1);
    float sc;
    if (s_valid[s]) {
      const float4* qp = (const float4*)(s_q + h * DKn);
      const float4* kp = (const float4*)(s_x + s * An + h * DKn);
      float acc = 0.f;
#pragma unroll
      for (int j = 0; j < 8; ++j) {
        const float4 qv = qp[j], kv = kp[j];
        acc += qv.x * kv.x + qv.y * kv.y + qv.z * kv.z + qv.w * kv.w;
      }
      sc = acc * scale;
    } else {
      sc = -__builtin_inff();
    }
    s_sc[it] = sc;
  }
  __syncthreads();

  // stage vW (reuse s_w, swizzled) — all threads participate
#pragma unroll
  for (int i = tid * 4; i < En * An; i += 256 * 4) {
    const int e = i >> 7, a = i & 127;
    const float4 w4 = *(const float4*)(vW + i);
    _Float16* dst = s_w + ((e >> 5) << 12);
    const int ek = e & 31;
    dst[swz_b(ek, a + 0)] = (_Float16)w4.x;
    dst[swz_b(ek, a + 1)] = (_Float16)w4.y;
    dst[swz_b(ek, a + 2)] = (_Float16)w4.z;
    dst[swz_b(ek, a + 3)] = (_Float16)w4.w;
  }

  // softmax per head (waves 0..3, shuffle reductions)
  if (tid < 128) {
    const int h = tid >> 5;
    float v0 = s_sc[h * KTOP + lane];
    float v1 = s_sc[h * KTOP + lane + 32];
    float v2 = s_sc[h * KTOP + lane + 64];
    float v3 = s_sc[h * KTOP + lane + 96];
    float m = fmaxf(fmaxf(v0, v1), fmaxf(v2, v3));
#pragma unroll
    for (int off = 16; off >= 1; off >>= 1) m = fmaxf(m, __shfl_xor(m, off, 32));
    const float e0 = __expf(v0 - m), e1 = __expf(v1 - m);
    const float e2 = __expf(v2 - m), e3 = __expf(v3 - m);
    float sum = e0 + e1 + e2 + e3;
#pragma unroll
    for (int off = 16; off >= 1; off >>= 1) sum += __shfl_xor(sum, off, 32);
    const float inv = 1.f / sum;
    s_sc[h * KTOP + lane]      = e0 * inv;
    s_sc[h * KTOP + lane + 32] = e1 * inv;
    s_sc[h * KTOP + lane + 64] = e2 * inv;
    s_sc[h * KTOP + lane + 96] = e3 * inv;
  }
  __syncthreads();

  // V = seq @ vW + vb via WMMA (overwrites s_x; k fully consumed)
#pragma unroll
  for (int ct = 0; ct < 8; ++ct) {
    v8f acc = {};
    acc = wmma16(a0, frag_b_sw(s_w + ct * 512, lane), acc);
    acc = wmma16(a1, frag_b_sw(s_w + 4096 + ct * 512, lane), acc);
    const int c = 16 * ct + nn;
    const float bias = vb[c];
#pragma unroll
    for (int i = 0; i < 8; ++i) s_x[(rowbase + i + 8 * hl) * An + c] = acc[i] + bias;
  }
  __syncthreads();

  // out[a] = sum_s attn[h(a)][s] * v[s][a]
  if (tid < An) {
    const int h = tid >> 5;
    float acc = 0.f;
    for (int s = 0; s < KTOP; ++s) acc += s_sc[h * KTOP + s] * s_x[s * An + tid];
    s_out[tid] = acc;
  }
  __syncthreads();

  // final 128->64 projection, write into concat feature buffer
  if (tid < ODn) {
    float acc = ob[tid];
#pragma unroll 8
    for (int a = 0; a < An; ++a) acc += s_out[a] * oW[a * ODn + tid];
    const int base = type ? (FOHn * En + FSPn * En + FMHn * ODn)   // 1472
                          : (FOHn * En + FSPn * En);               // 1216
    feats[(size_t)b * D0n + base + f * ODn + tid] = acc;
  }
}

// ---------------- K7: generic GEMM + bias + ReLU (f16 WMMA, f32 accum) -------
// C[M,N] = relu(A[M,K] @ Bm[K,N] + bias). 128x128 tile/WG, 8 waves in 4x2,
// each wave a 32x64 sub-tile = 2x4 WMMA accumulators. B staged swizzled.
__global__ __launch_bounds__(256) void k_gemm(const float* __restrict__ A,
                                              const float* __restrict__ Bm,
                                              const float* __restrict__ bias,
                                              float* __restrict__ C,
                                              int M, int N, int K, int doRelu) {
  __shared__ __align__(16) _Float16 As[128 * 32];   // row-major
  __shared__ __align__(16) _Float16 Bs[32 * 128];   // swizzled fragment order
  const int tid = threadIdx.x, lane = tid & 31, wave = tid >> 5;
  const int wr = wave & 3, wc = wave >> 2;
  const int row0 = blockIdx.y * 128, col0 = blockIdx.x * 128;

  const v8f zero = {};
  v8f acc[2][4];
#pragma unroll
  for (int mi = 0; mi < 2; ++mi)
#pragma unroll
    for (int ni = 0; ni < 4; ++ni) acc[mi][ni] = zero;

  for (int k0 = 0; k0 < K; k0 += 32) {
#pragma unroll
    for (int i = tid * 4; i < 128 * 32; i += 1024) {
      const int r = i >> 5, c = i & 31;
      const float4 a4 = *(const float4*)(A + (size_t)(row0 + r) * K + k0 + c);
      v4h h;
      h[0] = (_Float16)a4.x; h[1] = (_Float16)a4.y;
      h[2] = (_Float16)a4.z; h[3] = (_Float16)a4.w;
      *(v4h*)(As + i) = h;
    }
#pragma unroll
    for (int i = tid * 4; i < 32 * 128; i += 1024) {
      const int r = i >> 7, c = i & 127;
      const float4 w4 = *(const float4*)(Bm + (size_t)(k0 + r) * N + col0 + c);
      Bs[swz_b(r, c + 0)] = (_Float16)w4.x;
      Bs[swz_b(r, c + 1)] = (_Float16)w4.y;
      Bs[swz_b(r, c + 2)] = (_Float16)w4.z;
      Bs[swz_b(r, c + 3)] = (_Float16)w4.w;
    }
    __syncthreads();
#pragma unroll
    for (int mi = 0; mi < 2; ++mi) {
      const v16h a = frag_a(As + (wr * 32 + mi * 16) * 32, 32, lane);
#pragma unroll
      for (int ni = 0; ni < 4; ++ni) {
        const v16h bfr = frag_b_sw(Bs + (wc * 4 + ni) * 512, lane);
        acc[mi][ni] = wmma16(a, bfr, acc[mi][ni]);
      }
    }
    __syncthreads();
  }

  const int hl = lane >> 4, nn = lane & 15;
#pragma unroll
  for (int mi = 0; mi < 2; ++mi)
#pragma unroll
    for (int ni = 0; ni < 4; ++ni) {
      const int gc = col0 + wc * 64 + ni * 16 + nn;
      const float bval = bias[gc];
#pragma unroll
      for (int i = 0; i < 8; ++i) {
        const int gr = row0 + wr * 32 + mi * 16 + i + 8 * hl;
        float v = acc[mi][ni][i] + bval;
        if (doRelu) v = fmaxf(v, 0.f);
        C[(size_t)gr * N + gc] = v;
      }
    }
}

// ---------------- K8: final 256->1 + sigmoid, pack (sigmoid | logits) --------
__global__ void k_final(const float* __restrict__ x3, const float* __restrict__ outW,
                        const float* __restrict__ outb, float* __restrict__ out) {
  const int b = blockIdx.x * 256 + threadIdx.x;
  if (b >= Bn) return;
  const float4* xp = (const float4*)(x3 + (size_t)b * 256);
  const float4* wp = (const float4*)outW;
  float acc = outb[0];
#pragma unroll 8
  for (int j = 0; j < 64; ++j) {
    const float4 xv = xp[j], wv = wp[j];
    acc += xv.x * wv.x + xv.y * wv.y + xv.z * wv.z + xv.w * wv.w;
  }
  out[b] = 1.f / (1.f + __expf(-acc));
  out[Bn + b] = acc;
}

// ---------------- host launcher ----------------
extern "C" void kernel_launch(void* const* d_in, const int* in_sizes, int n_in,
                              void* d_out, int out_size, void* d_ws, size_t ws_size,
                              hipStream_t stream) {
  (void)in_sizes; (void)n_in; (void)out_size; (void)ws_size;
  const float* ohT  = (const float*)d_in[0];
  const float* mhT  = (const float*)d_in[1];
  const float* spT  = (const float*)d_in[2];
  const float* hw   = (const float*)d_in[3];
  const float* sqW  = (const float*)d_in[4];
  const float* sqb  = (const float*)d_in[5];
  const float* skW  = (const float*)d_in[6];
  const float* skb  = (const float*)d_in[7];
  const float* svW  = (const float*)d_in[8];
  const float* svb  = (const float*)d_in[9];
  const float* soW  = (const float*)d_in[10];
  const float* sob  = (const float*)d_in[11];
  const float* lqW  = (const float*)d_in[12];
  const float* lqb  = (const float*)d_in[13];
  const float* lkW  = (const float*)d_in[14];
  const float* lkb  = (const float*)d_in[15];
  const float* lvW  = (const float*)d_in[16];
  const float* lvb  = (const float*)d_in[17];
  const float* loW  = (const float*)d_in[18];
  const float* lob  = (const float*)d_in[19];
  const float* W0   = (const float*)d_in[20];
  const float* b0   = (const float*)d_in[21];
  const float* W1   = (const float*)d_in[22];
  const float* b1   = (const float*)d_in[23];
  const float* W2   = (const float*)d_in[24];
  const float* b2   = (const float*)d_in[25];
  const float* outW = (const float*)d_in[26];
  const float* outb = (const float*)d_in[27];
  const int* ohIds  = (const int*)d_in[28];
  const int* mhIds  = (const int*)d_in[29];
  const int* spIds  = (const int*)d_in[30];

  char* ws = (char*)d_ws;
  unsigned int* mhCodes  = (unsigned int*)(ws + 0);          // 4*100001*4  = 1,600,016
  unsigned int* tgtCodes = (unsigned int*)(ws + 1600512);    // 2048*4      = 8,192
  int*   topkIdx = (int*)  (ws + 1608704);                   // 4*512*128*4 = 1,048,576
  float* feats   = (float*)(ws + 2657280);                   // 512*1728*4  = 3,538,944
  float* x1      = (float*)(ws + 6196224);                   // 512*1024*4  = 2,097,152
  float* x2      = (float*)(ws + 8293376);                   // 512*512*4   = 1,048,576
  float* x3      = (float*)(ws + 9341952);                   // 512*256*4   =   524,288

  // 1) LSH codes (vocab rows + targets)
  k_hash_mh<<<(NROWS_MH + 7) / 8, 256, 0, stream>>>(mhT, hw, mhCodes);
  k_hash_tgt<<<(FMHn * Bn) / 8, 256, 0, stream>>>(ohT, hw, ohIds, tgtCodes);

  // 2) feature-buffer gathers
  k_oh_feats<<<(Bn * FOHn * En + 255) / 256, 256, 0, stream>>>(ohT, ohIds, feats);
  k_sp_feats<<<(FSPn * Bn * En + 255) / 256, 256, 0, stream>>>(spT, spIds, feats);

  // 3) exact stable top-128 retrieval
  k_topk<<<FMHn * Bn, 256, 0, stream>>>(mhIds, mhCodes, tgtCodes, topkIdx);

  // 4) short + long attention (grid bit 11 selects type), 96 KB dynamic LDS
  k_attn<<<2 * FMHn * Bn, 256, 98304, stream>>>(
      ohT, mhT, ohIds, mhIds, topkIdx,
      sqW, sqb, skW, skb, svW, svb, soW, sob,
      lqW, lqb, lkW, lkb, lvW, lvb, loW, lob, feats);

  // 5) MLP: 1728 -> 1024 -> 512 -> 256 (WMMA), then 256 -> 1 + sigmoid
  k_gemm<<<dim3(8, 4), 256, 0, stream>>>(feats, W0, b0, x1, Bn, 1024, D0n, 1);
  k_gemm<<<dim3(4, 4), 256, 0, stream>>>(x1, W1, b1, x2, Bn, 512, 1024, 1);
  k_gemm<<<dim3(2, 4), 256, 0, stream>>>(x2, W2, b2, x3, Bn, 256, 512, 1);
  k_final<<<(Bn + 255) / 256, 256, 0, stream>>>(x3, outW, outb, (float*)d_out);
}
